// Temporal_Forward_58110907515332
// MI455X (gfx1250) — matrix operand order repaired and checked by reading, not compiled, not run on hardware
//
#include <hip/hip_runtime.h>
#include <hip/hip_bf16.h>
#include <math.h>

typedef _Float16 v8h  __attribute__((ext_vector_type(8)));
typedef _Float16 v16h __attribute__((ext_vector_type(16)));
typedef float    v8f  __attribute__((ext_vector_type(8)));

#define T_   243
#define C_   544
#define BS_  3
#define NB_  81
#define K_   6
#define H_   8
#define DH_  68
#define HID_ 1088
#define EPS_ 1e-6f

// ---------------------------------------------------------------------------
// f32 -> f16 weight conversion (weights stay resident across GEMMs; tiny)
// ---------------------------------------------------------------------------
__global__ __launch_bounds__(256) void cvt_f32_f16(const float* __restrict__ src,
                                                   _Float16* __restrict__ dst, int n) {
  int i = blockIdx.x * 256 + threadIdx.x;
  if (i < n) dst[i] = (_Float16)src[i];
}

// ---------------------------------------------------------------------------
// Stage A: single pass over x. y[b,n] = mean over (s,c) of shifted (x+pos).
// shift: c<272 -> (t>0 ? x[t-1]+pos[t-1] : 0); c>=272 -> x[t]+pos[t]
// ---------------------------------------------------------------------------
__global__ __launch_bounds__(256) void block_mean_kernel(const float* __restrict__ x,
                                                         const float* __restrict__ pos,
                                                         float* __restrict__ y, int B) {
  int n = blockIdx.x % NB_;
  int b = blockIdx.x / NB_;
  int tid = threadIdx.x;
  float s = 0.f;
  for (int i = tid; i < BS_ * C_; i += 256) {
    int ss = i / C_, c = i % C_;
    int t = n * BS_ + ss;
    float v;
    if (c < C_ / 2) {
      v = (t > 0) ? x[((size_t)b * T_ + (t - 1)) * C_ + c] + pos[(size_t)(t - 1) * C_ + c] : 0.f;
    } else {
      v = x[((size_t)b * T_ + t) * C_ + c] + pos[(size_t)t * C_ + c];
    }
    s += v;
  }
  __shared__ float sh[256];
  sh[tid] = s; __syncthreads();
  for (int off = 128; off > 0; off >>= 1) {
    if (tid < off) sh[tid] += sh[tid + off];
    __syncthreads();
  }
  if (tid == 0) y[(size_t)b * NB_ + n] = sh[0] / (float)(BS_ * C_);
}

// ---------------------------------------------------------------------------
// Stage B: gating MLP (81x81 twice) + top-6 selection. One block per batch.
// ---------------------------------------------------------------------------
__global__ __launch_bounds__(128) void mlp_topk_kernel(const float* __restrict__ y,
                                                       const float* __restrict__ fc1,
                                                       const float* __restrict__ fc2,
                                                       int* __restrict__ idx) {
  int b = blockIdx.x;
  int tid = threadIdx.x;
  __shared__ float yv[NB_], h1[NB_], z[NB_];
  if (tid < NB_) yv[tid] = y[(size_t)b * NB_ + tid];
  __syncthreads();
  if (tid < NB_) {
    float h = 0.f;
    for (int j = 0; j < NB_; ++j) h += fc1[tid * NB_ + j] * yv[j];
    h1[tid] = fmaxf(h, 0.f);
  }
  __syncthreads();
  if (tid < NB_) {
    float h = 0.f;
    for (int j = 0; j < NB_; ++j) h += fc2[tid * NB_ + j] * h1[j];
    z[tid] = 1.f / (1.f + expf(-h));
  }
  __syncthreads();
  if (tid == 0) {
    bool used[NB_];
    for (int i = 0; i < NB_; ++i) used[i] = false;
    for (int k = 0; k < K_; ++k) {
      float best = -3.4e38f; int bi = 0;
      for (int i = 0; i < NB_; ++i)
        if (!used[i] && z[i] > best) { best = z[i]; bi = i; }
      used[bi] = true;
      idx[b * K_ + k] = bi;
    }
  }
}

// ---------------------------------------------------------------------------
// Stage A': gather the K selected blocks, recomputing shift+mean from x.
// ---------------------------------------------------------------------------
__global__ __launch_bounds__(128) void gather_kernel(const float* __restrict__ x,
                                                     const float* __restrict__ pos,
                                                     const int* __restrict__ idx,
                                                     float* __restrict__ t) {
  int id = blockIdx.x;           // b*K + k
  int b = id / K_;
  int n = idx[id];
  for (int c = threadIdx.x; c < C_; c += 128) {
    float acc = 0.f;
    for (int ss = 0; ss < BS_; ++ss) {
      int tt = n * BS_ + ss;
      float v;
      if (c < C_ / 2) {
        v = (tt > 0) ? x[((size_t)b * T_ + (tt - 1)) * C_ + c] + pos[(size_t)(tt - 1) * C_ + c] : 0.f;
      } else {
        v = x[((size_t)b * T_ + tt) * C_ + c] + pos[(size_t)tt * C_ + c];
      }
      acc += v;
    }
    t[(size_t)id * C_ + c] = acc / (float)BS_;
  }
}

// ---------------------------------------------------------------------------
// LayerNorm over rows of length L, output f16 (GEMM A operand). 1 block/row.
// ---------------------------------------------------------------------------
__global__ __launch_bounds__(128) void ln_rows_f16(const float* __restrict__ X,
                                                   const float* __restrict__ g,
                                                   const float* __restrict__ bb,
                                                   _Float16* __restrict__ out, int L) {
  int row = blockIdx.x, tid = threadIdx.x;
  const float* xr = X + (size_t)row * L;
  float s = 0.f, s2 = 0.f;
  for (int c = tid; c < L; c += 128) { float v = xr[c]; s += v; s2 += v * v; }
  __shared__ float sh[128], sh2[128];
  sh[tid] = s; sh2[tid] = s2; __syncthreads();
  for (int off = 64; off > 0; off >>= 1) {
    if (tid < off) { sh[tid] += sh[tid + off]; sh2[tid] += sh2[tid + off]; }
    __syncthreads();
  }
  float mean = sh[0] / (float)L;
  float var  = sh2[0] / (float)L - mean * mean;
  float rs = rsqrtf(var + EPS_);
  for (int c = tid; c < L; c += 128)
    out[(size_t)row * L + c] = (_Float16)((xr[c] - mean) * rs * g[c] + bb[c]);
}

// ---------------------------------------------------------------------------
// Core GEMM via v_wmma_f32_16x16x32_f16, 2x2 16x16 tiles per wave (32x32 out).
//   Out[M,N] = epilogue( A[M,Kd] * W[N,Kd]^T + bias[N] )
// mode 0: store f32   mode 1: f32 residual add   mode 2: exact GELU -> f16
// A frag reused across 2 N-tiles, B frag reused across 2 M-tiles:
// 4 WMMA per 8 b128 loads per k-step.
// ---------------------------------------------------------------------------
#define MT 2
#define NT 2
__global__ __launch_bounds__(256) void gemm_f16_wmma(const _Float16* __restrict__ A,
                                                     const _Float16* __restrict__ W,
                                                     const float* __restrict__ bias,
                                                     void* __restrict__ Out,
                                                     int M, int N, int Kd, int mode) {
  const int lane = threadIdx.x & 31;
  const int Ntg = N / (16 * NT);
  const int Mtg = M / (16 * MT);
  int wid = blockIdx.x * 8 + (threadIdx.x >> 5);
  if (wid >= Mtg * Ntg) return;
  const int m0 = (wid / Ntg) * (16 * MT);
  const int n0 = (wid % Ntg) * (16 * NT);

  v8f c00 = {}, c01 = {}, c10 = {}, c11 = {};

  const int mr = lane & 15;
  const int aoff = (lane & 16) ? 16 : 0;   // A: hi lanes take K+16..31
  const int koffB = (lane & 16) ? 16 : 0;  // B: hi lanes take K+16..31
  const _Float16* A0 = A + (size_t)(m0 + mr) * Kd + aoff;
  const _Float16* A1 = A + (size_t)(m0 + 16 + mr) * Kd + aoff;
  const _Float16* B0 = W + (size_t)(n0 + mr) * Kd + koffB;
  const _Float16* B1 = W + (size_t)(n0 + 16 + mr) * Kd + koffB;

  for (int k = 0; k < Kd; k += 32) {
    v16h a0, a1;
    *((v8h*)&a0)       = *(const v8h*)(A0 + k);
    *(((v8h*)&a0) + 1) = *(const v8h*)(A0 + k + 8);
    *((v8h*)&a1)       = *(const v8h*)(A1 + k);
    *(((v8h*)&a1) + 1) = *(const v8h*)(A1 + k + 8);
    v16h b0 = *(const v16h*)(B0 + k);
    v16h b1 = *(const v16h*)(B1 + k);
    if (k + 64 < Kd) {
      __builtin_prefetch(A0 + k + 64, 0, 1);
      __builtin_prefetch(A1 + k + 64, 0, 1);
    }
    c00 = __builtin_amdgcn_wmma_f32_16x16x32_f16(false, a0, false, b0, (short)0, c00, false, false);
    c01 = __builtin_amdgcn_wmma_f32_16x16x32_f16(false, a0, false, b1, (short)0, c01, false, false);
    c10 = __builtin_amdgcn_wmma_f32_16x16x32_f16(false, a1, false, b0, (short)0, c10, false, false);
    c11 = __builtin_amdgcn_wmma_f32_16x16x32_f16(false, a1, false, b1, (short)0, c11, false, false);
  }

  // D layout: col = n-tile + (lane&15); row r -> M = r (+8 for hi lanes)
  const int rbase = (lane & 16) ? 8 : 0;
  v8f acc[MT][NT] = { { c00, c01 }, { c10, c11 } };
#pragma unroll
  for (int i = 0; i < MT; ++i) {
    int mb = m0 + i * 16 + rbase;
#pragma unroll
    for (int j = 0; j < NT; ++j) {
      int col = n0 + j * 16 + (lane & 15);
      float bv = bias ? bias[col] : 0.f;
#pragma unroll
      for (int r = 0; r < 8; ++r) {
        float val = acc[i][j][r] + bv;
        size_t o = (size_t)(mb + r) * N + col;
        if (mode == 0)      ((float*)Out)[o] = val;
        else if (mode == 1) ((float*)Out)[o] += val;
        else ((_Float16*)Out)[o] =
            (_Float16)(0.5f * val * (1.0f + erff(val * 0.70710678118654752f)));
      }
    }
  }
}

// ---------------------------------------------------------------------------
// Attention: K_=6 tokens, per (batch, head) block. q/k/v staged into LDS via
// gfx1250 async global->LDS copies (ASYNCcnt), then tiny softmax.
// qkv row layout: q @ head*68+d, k @ 544+..., v @ 1088+...
// ---------------------------------------------------------------------------
__global__ __launch_bounds__(128) void attn_kernel(const float* __restrict__ qkv,
                                                   _Float16* __restrict__ o16) {
  __shared__ float smem[3 * K_ * DH_];   // [which][tok][d]
  __shared__ float sc[K_][K_];
  int b = blockIdx.x / H_;
  int head = blockIdx.x % H_;
  int tid = threadIdx.x;

  // async copy 3*6*68 dwords: global -> LDS, tracked by ASYNCcnt
  for (int i = tid; i < 3 * K_ * DH_; i += 128) {
    int which = i / (K_ * DH_);
    int r = i % (K_ * DH_);
    int tok = r / DH_, d = r % DH_;
    const float* gsrc = qkv + ((size_t)b * K_ + tok) * (3 * C_) + which * C_ + head * DH_ + d;
    unsigned lds_off = (unsigned)(size_t)(smem + i);       // low 32 bits = LDS byte addr
    unsigned long long ga = (unsigned long long)(size_t)gsrc;
    asm volatile("global_load_async_to_lds_b32 %0, %1, off"
                 :: "v"(lds_off), "v"(ga) : "memory");
  }
  asm volatile("s_wait_asynccnt 0x0" ::: "memory");
  __syncthreads();

  const float* q  = smem;
  const float* kk = smem + K_ * DH_;
  const float* vv = smem + 2 * K_ * DH_;

  if (tid < K_ * K_) {
    int tn = tid / K_, tm = tid % K_;
    float s = 0.f;
    for (int d = 0; d < DH_; ++d) s += q[tn * DH_ + d] * kk[tm * DH_ + d];
    sc[tn][tm] = s * 0.12126781251816650f;  // 68^-0.5
  }
  __syncthreads();
  if (tid < K_) {
    float mx = -3.4e38f;
    for (int m = 0; m < K_; ++m) mx = fmaxf(mx, sc[tid][m]);
    float sum = 0.f;
    for (int m = 0; m < K_; ++m) { float e = expf(sc[tid][m] - mx); sc[tid][m] = e; sum += e; }
    float inv = 1.f / sum;
    for (int m = 0; m < K_; ++m) sc[tid][m] *= inv;
  }
  __syncthreads();
  for (int i = tid; i < K_ * DH_; i += 128) {
    int tok = i / DH_, d = i % DH_;
    float o = 0.f;
    for (int m = 0; m < K_; ++m) o += sc[tok][m] * vv[m * DH_ + d];
    o16[((size_t)b * K_ + tok) * C_ + head * DH_ + d] = (_Float16)o;
  }
}

// ---------------------------------------------------------------------------
// Final: per-batch, LN each of the K rows, weighted-sum with wm2_w, + wm2_b.
// ---------------------------------------------------------------------------
__global__ __launch_bounds__(128) void final_kernel(const float* __restrict__ t,
                                                    const float* __restrict__ g,
                                                    const float* __restrict__ bb,
                                                    const float* __restrict__ w,
                                                    const float* __restrict__ wb,
                                                    float* __restrict__ out) {
  int b = blockIdx.x, tid = threadIdx.x;
  __shared__ float acc[C_];
  __shared__ float sh[128], sh2[128];
  for (int c = tid; c < C_; c += 128) acc[c] = 0.f;
  __syncthreads();
  for (int k = 0; k < K_; ++k) {
    const float* row = t + ((size_t)b * K_ + k) * C_;
    float s = 0.f, s2 = 0.f;
    for (int c = tid; c < C_; c += 128) { float v = row[c]; s += v; s2 += v * v; }
    sh[tid] = s; sh2[tid] = s2; __syncthreads();
    for (int off = 64; off > 0; off >>= 1) {
      if (tid < off) { sh[tid] += sh[tid + off]; sh2[tid] += sh2[tid + off]; }
      __syncthreads();
    }
    float mean = sh[0] / (float)C_;
    float rs = rsqrtf(sh2[0] / (float)C_ - mean * mean + EPS_);
    float wk = w[k];
    for (int c = tid; c < C_; c += 128)
      acc[c] += wk * ((row[c] - mean) * rs * g[c] + bb[c]);
    __syncthreads();
  }
  for (int c = tid; c < C_; c += 128) out[(size_t)b * C_ + c] = acc[c] + wb[0];
}

// ---------------------------------------------------------------------------
// Host side
// ---------------------------------------------------------------------------
extern "C" void kernel_launch(void* const* d_in, const int* in_sizes, int n_in,
                              void* d_out, int out_size, void* d_ws, size_t ws_size,
                              hipStream_t stream) {
  const float* x       = (const float*)d_in[0];
  const float* pos     = (const float*)d_in[1];
  const float* fc1_w   = (const float*)d_in[2];
  const float* fc2_w   = (const float*)d_in[3];
  const float* ln1_g   = (const float*)d_in[4];
  const float* ln1_b   = (const float*)d_in[5];
  const float* qkv_w   = (const float*)d_in[6];
  const float* qkv_b   = (const float*)d_in[7];
  const float* proj_w  = (const float*)d_in[8];
  const float* proj_b  = (const float*)d_in[9];
  const float* ln2_g   = (const float*)d_in[10];
  const float* ln2_b   = (const float*)d_in[11];
  const float* mfc1_w  = (const float*)d_in[12];
  const float* mfc1_b  = (const float*)d_in[13];
  const float* mfc2_w  = (const float*)d_in[14];
  const float* mfc2_b  = (const float*)d_in[15];
  const float* lnf_g   = (const float*)d_in[16];
  const float* lnf_b   = (const float*)d_in[17];
  const float* wm2_w   = (const float*)d_in[18];
  const float* wm2_b   = (const float*)d_in[19];
  float* out = (float*)d_out;

  const int B = in_sizes[0] / (T_ * C_);
  const int M = B * K_;            // token rows (512*6 = 3072, multiple of 32)
  const int Mtg = M / (16 * MT);

  // ---- workspace carving (256B aligned) ----
  char* base = (char*)d_ws;
  size_t off = 0;
  auto carve = [&](size_t bytes) -> void* {
    void* p = base + off;
    off = (off + bytes + 255) & ~(size_t)255;
    return p;
  };
  float*     y_ws   = (float*)    carve((size_t)B * NB_ * 4);
  int*       idx_ws = (int*)      carve((size_t)B * K_ * 4);
  float*     t_ws   = (float*)    carve((size_t)M * C_ * 4);
  float*     qkvbuf = (float*)    carve((size_t)M * 3 * C_ * 4);
  _Float16*  a16    = (_Float16*) carve((size_t)M * C_ * 2);
  _Float16*  o16    = (_Float16*) carve((size_t)M * C_ * 2);
  _Float16*  g16    = (_Float16*) carve((size_t)M * HID_ * 2);
  _Float16*  qkvw16 = (_Float16*) carve((size_t)2 * 3 * C_ * C_ * 2);
  _Float16*  projw16= (_Float16*) carve((size_t)2 * C_ * C_ * 2);
  _Float16*  m1w16  = (_Float16*) carve((size_t)2 * HID_ * C_ * 2);
  _Float16*  m2w16  = (_Float16*) carve((size_t)2 * C_ * HID_ * 2);
  (void)ws_size; (void)n_in; (void)out_size;

  // ---- weight conversion (f32 -> f16) ----
  {
    int n;
    n = 2 * 3 * C_ * C_;  cvt_f32_f16<<<(n + 255) / 256, 256, 0, stream>>>(qkv_w,  qkvw16, n);
    n = 2 * C_ * C_;      cvt_f32_f16<<<(n + 255) / 256, 256, 0, stream>>>(proj_w, projw16, n);
    n = 2 * HID_ * C_;    cvt_f32_f16<<<(n + 255) / 256, 256, 0, stream>>>(mfc1_w, m1w16, n);
    n = 2 * C_ * HID_;    cvt_f32_f16<<<(n + 255) / 256, 256, 0, stream>>>(mfc2_w, m2w16, n);
  }

  // ---- stage A: single pass over x -> y ----
  block_mean_kernel<<<B * NB_, 256, 0, stream>>>(x, pos, y_ws, B);
  // ---- stage B: gating MLP + top-6 ----
  mlp_topk_kernel<<<B, 128, 0, stream>>>(y_ws, fc1_w, fc2_w, idx_ws);
  // ---- gather selected blocks (recompute shift+mean) ----
  gather_kernel<<<B * K_, 128, 0, stream>>>(x, pos, idx_ws, t_ws);

  // ---- transformer, DEPTH=2 ----
  auto gemm_grid = [&](int N) { return (Mtg * (N / (16 * NT)) + 7) / 8; };
  for (int i = 0; i < 2; ++i) {
    ln_rows_f16<<<M, 128, 0, stream>>>(t_ws, ln1_g + i * C_, ln1_b + i * C_, a16, C_);
    gemm_f16_wmma<<<gemm_grid(3 * C_), 256, 0, stream>>>(
        a16, qkvw16 + (size_t)i * 3 * C_ * C_, qkv_b + i * 3 * C_, qkvbuf,
        M, 3 * C_, C_, 0);
    attn_kernel<<<B * H_, 128, 0, stream>>>(qkvbuf, o16);
    gemm_f16_wmma<<<gemm_grid(C_), 256, 0, stream>>>(
        o16, projw16 + (size_t)i * C_ * C_, proj_b + i * C_, t_ws,
        M, C_, C_, 1);
    ln_rows_f16<<<M, 128, 0, stream>>>(t_ws, ln2_g + i * C_, ln2_b + i * C_, a16, C_);
    gemm_f16_wmma<<<gemm_grid(HID_), 256, 0, stream>>>(
        a16, m1w16 + (size_t)i * HID_ * C_, mfc1_b + i * HID_, g16,
        M, HID_, C_, 2);
    gemm_f16_wmma<<<gemm_grid(C_), 256, 0, stream>>>(
        g16, m2w16 + (size_t)i * C_ * HID_, mfc2_b + i * C_, t_ws,
        M, C_, HID_, 1);
  }

  // ---- final LN + weighted sum over K ----
  final_kernel<<<B, 128, 0, stream>>>(t_ws, lnf_g, lnf_b, wm2_w, wm2_b, out);
}
